// Attention_23338852286907
// MI455X (gfx1250) — matrix operand order repaired and checked by reading, not compiled, not run on hardware
//
#include <hip/hip_runtime.h>
#include <hip/hip_bf16.h>

#define B_ 8
#define S_ 2048
#define D_ 1024
#define MROWS (B_*S_)          // 16384 total rows
#define CHUNK 512              // key chunk for flash attention
#define SCALE 0.02209708691207961f  // 1/sqrt(2048)  (reference scales by sqrt(S))

typedef __attribute__((ext_vector_type(16))) __bf16 v16bf;
typedef __attribute__((ext_vector_type(8)))  float  v8f;

union ABOp { v16bf v; uint4 q[2]; };

__device__ __forceinline__ v8f wmma_bf16(const ABOp& a, const ABOp& b, v8f c) {
    // 8 args: (neg_a, A, neg_b, B, c_mod, C, reuse_a, reuse_b)
    return __builtin_amdgcn_wmma_f32_16x16x32_bf16(
        false, a.v, false, b.v, (short)0, c, false, false);
}

// A operand: lane holds K [k0+16h,+16)  (h = lane>>4)
#define LD_A(dst, p) { (dst).q[0] = *(const uint4*)(p); \
                       (dst).q[1] = *(const uint4*)((p) + 8); }
// B operand: lane holds 32 contiguous bytes of its column's K range
#define LD_B(dst, p) { (dst).q[0] = *(const uint4*)(p); \
                       (dst).q[1] = *(const uint4*)((p) + 8); }

// Quad-tile K-loop: ONE A stream shared by four B tiles.
//  - Trip count laundered through an SGPR asm barrier: the optimizer cannot
//    fully unroll, so operand addresses vary with the induction pointers and
//    LICM/GVN cannot hoist loads out of this loop or any enclosing loop
//    (the round-2..4 "all A operands live at once" register explosion).
//  - Explicit even/odd ping-pong, two k-steps per iteration: no rotation
//    copies, distance-1 prefetch, bounded live set (~2x5 operand regs + accs).
#define QUAD_K_LOOP(aB, b0B, b1B, b2B, b3B, KLEN, acc0, acc1, acc2, acc3)     \
  {                                                                           \
    const __bf16 *ap_ = (aB), *b0p_ = (b0B), *b1p_ = (b1B),                   \
                 *b2p_ = (b2B), *b3p_ = (b3B);                                \
    ABOp aE, b0E, b1E, b2E, b3E, aO, b0O, b1O, b2O, b3O;                      \
    LD_A(aE, ap_);                                                            \
    LD_B(b0E, b0p_); LD_B(b1E, b1p_); LD_B(b2E, b2p_); LD_B(b3E, b3p_);       \
    int it_ = (KLEN) / 64 - 1;                                                \
    asm volatile("" : "+s"(it_));   /* opaque trip count: keep loop rolled */ \
    for (int i_ = 0; i_ < it_; ++i_) {                                        \
      LD_A(aO, ap_ + 32);                                                     \
      LD_B(b0O, b0p_ + 32); LD_B(b1O, b1p_ + 32);                             \
      LD_B(b2O, b2p_ + 32); LD_B(b3O, b3p_ + 32);                             \
      acc0 = wmma_bf16(aE, b0E, acc0);                                        \
      acc1 = wmma_bf16(aE, b1E, acc1);                                        \
      acc2 = wmma_bf16(aE, b2E, acc2);                                        \
      acc3 = wmma_bf16(aE, b3E, acc3);                                        \
      LD_A(aE, ap_ + 64);                                                     \
      LD_B(b0E, b0p_ + 64); LD_B(b1E, b1p_ + 64);                             \
      LD_B(b2E, b2p_ + 64); LD_B(b3E, b3p_ + 64);                             \
      acc0 = wmma_bf16(aO, b0O, acc0);                                        \
      acc1 = wmma_bf16(aO, b1O, acc1);                                        \
      acc2 = wmma_bf16(aO, b2O, acc2);                                        \
      acc3 = wmma_bf16(aO, b3O, acc3);                                        \
      ap_ += 64; b0p_ += 64; b1p_ += 64; b2p_ += 64; b3p_ += 64;              \
    }                                                                         \
    LD_A(aO, ap_ + 32);                                                       \
    LD_B(b0O, b0p_ + 32); LD_B(b1O, b1p_ + 32);                               \
    LD_B(b2O, b2p_ + 32); LD_B(b3O, b3p_ + 32);                               \
    acc0 = wmma_bf16(aE, b0E, acc0);                                          \
    acc1 = wmma_bf16(aE, b1E, acc1);                                          \
    acc2 = wmma_bf16(aE, b2E, acc2);                                          \
    acc3 = wmma_bf16(aE, b3E, acc3);                                          \
    acc0 = wmma_bf16(aO, b0O, acc0);                                          \
    acc1 = wmma_bf16(aO, b1O, acc1);                                          \
    acc2 = wmma_bf16(aO, b2O, acc2);                                          \
    acc3 = wmma_bf16(aO, b3O, acc3);                                          \
  }

// ---------------------------------------------------------------------------
// K0: convert the three f32 weight matrices [D,D] to bf16 (concatenated)
// ---------------------------------------------------------------------------
__global__ __launch_bounds__(256) void k_convert_w(
    const float* __restrict__ qw, const float* __restrict__ kw,
    const float* __restrict__ vw, __bf16* __restrict__ wb) {
    const long n = (long)D_ * D_;
    long base = ((long)blockIdx.x * 256 + threadIdx.x) * 4;
    const float* src;
    long off;
    if (base < n)          { src = qw; off = base; }
    else if (base < 2 * n) { src = kw; off = base - n; }
    else                   { src = vw; off = base - 2 * n; }
    float4 f = *(const float4*)(src + off);
    wb[base + 0] = (__bf16)f.x;
    wb[base + 1] = (__bf16)f.y;
    wb[base + 2] = (__bf16)f.z;
    wb[base + 3] = (__bf16)f.w;
}

// ---------------------------------------------------------------------------
// K1: QKV projection.  y = X @ W^T + b  (NT GEMM, both K-contiguous).
// grid = (MROWS/16, D/512), block = 256 (8 waves). Wave -> four 16-col tiles.
// Q,K stored [B*S, D] bf16; V stored transposed Vt[b][d][s] bf16.
// ---------------------------------------------------------------------------
__global__ __launch_bounds__(256) void k_qkv(
    const float* __restrict__ X, const __bf16* __restrict__ Wb,
    const float* __restrict__ bq, const float* __restrict__ bk,
    const float* __restrict__ bv,
    __bf16* __restrict__ Qb, __bf16* __restrict__ Kb, __bf16* __restrict__ Vtb) {
    __shared__ __bf16 Xs[16 * D_];                       // 32 KB
    const int tid = threadIdx.x;
    const long row0 = (long)blockIdx.x * 16;
    const float* xsrc = X + row0 * D_;
    for (int i = tid; i < 16 * D_ / 4; i += 256) {       // f32 -> bf16 stage
        float4 f = ((const float4*)xsrc)[i];
        __bf16* d = Xs + (long)i * 4;
        d[0] = (__bf16)f.x; d[1] = (__bf16)f.y;
        d[2] = (__bf16)f.z; d[3] = (__bf16)f.w;
    }
    __syncthreads();
    const int wave = tid >> 5, lane = tid & 31, half = lane >> 4, l15 = lane & 15;
    const int n0 = blockIdx.y * 512 + wave * 64;   // tiles n0, +16, +32, +48
    const __bf16* aB = Xs + l15 * D_ + half * 16;
    for (int mat = 0; mat < 3; ++mat) {
        const __bf16* W = Wb + (long)mat * D_ * D_;
        const float* bias_p = (mat == 0) ? bq : ((mat == 1) ? bk : bv);
        const __bf16* b0B = W + (long)(n0 + l15) * D_ + half * 16;
        const __bf16* b1B = W + (long)(n0 + 16 + l15) * D_ + half * 16;
        const __bf16* b2B = W + (long)(n0 + 32 + l15) * D_ + half * 16;
        const __bf16* b3B = W + (long)(n0 + 48 + l15) * D_ + half * 16;
        if (mat < 2) {  // warm L2 for the next matrix's weight rows
            __builtin_prefetch(W + (long)D_ * D_ + (long)(n0 + l15) * D_, 0, 1);
        }
        v8f a0 = {}, a1 = {}, a2 = {}, a3 = {};
        QUAD_K_LOOP(aB, b0B, b1B, b2B, b3B, D_, a0, a1, a2, a3);
        float bi0 = bias_p[n0 + l15];
        float bi1 = bias_p[n0 + 16 + l15];
        float bi2 = bias_p[n0 + 32 + l15];
        float bi3 = bias_p[n0 + 48 + l15];
        if (mat < 2) {
            __bf16* out = (mat == 0) ? Qb : Kb;
            #pragma unroll
            for (int r = 0; r < 8; ++r) {
                long row = row0 + r + 8 * half;
                out[row * D_ + n0 + l15]      = (__bf16)(a0[r] + bi0);
                out[row * D_ + n0 + 16 + l15] = (__bf16)(a1[r] + bi1);
                out[row * D_ + n0 + 32 + l15] = (__bf16)(a2[r] + bi2);
                out[row * D_ + n0 + 48 + l15] = (__bf16)(a3[r] + bi3);
            }
        } else {
            #pragma unroll
            for (int r = 0; r < 8; ++r) {
                long row = row0 + r + 8 * half;
                long bb = row >> 11, s = row & (S_ - 1);
                long base = bb * D_;
                Vtb[(base + n0 + l15) * (long)S_ + s]      = (__bf16)(a0[r] + bi0);
                Vtb[(base + n0 + 16 + l15) * (long)S_ + s] = (__bf16)(a1[r] + bi1);
                Vtb[(base + n0 + 32 + l15) * (long)S_ + s] = (__bf16)(a2[r] + bi2);
                Vtb[(base + n0 + 48 + l15) * (long)S_ + s] = (__bf16)(a3[r] + bi3);
            }
        }
    }
}

// ---------------------------------------------------------------------------
// K2: flash attention (context only).  grid = (S/16 qtiles, B), block = 256.
// Online softmax over key chunks of 512.  Wave: 64 keys in ONE quad K-loop
// for scores; 128 d-columns as 2 quad loops for P.V.
// ---------------------------------------------------------------------------
__global__ __launch_bounds__(256) void k_attn(
    const __bf16* __restrict__ Qb, const __bf16* __restrict__ Kb,
    const __bf16* __restrict__ Vtb, float* __restrict__ ctx) {
    __shared__ float  Ssc[16 * CHUNK];     // 32 KB raw scaled scores
    __shared__ __bf16 Ps[16 * CHUNK];      // 16 KB exp() probabilities
    __shared__ float  red[16 * 16];
    __shared__ float  rowMax[16], rowSum[16], rowScale[16];
    const int tid = threadIdx.x, wave = tid >> 5, lane = tid & 31;
    const int half = lane >> 4, l15 = lane & 15;
    const int batch = blockIdx.y;
    const long q0 = (long)batch * S_ + blockIdx.x * 16;   // global query row 0
    const int rrow = tid >> 4, seg = tid & 15;            // softmax partition
    if (tid < 16) { rowMax[tid] = -3.0e38f; rowSum[tid] = 0.f; }
    __syncthreads();
    v8f accO[8];
    #pragma unroll
    for (int t = 0; t < 8; ++t) accO[t] = (v8f){};
    const __bf16* aQ = Qb + (q0 + l15) * D_ + half * 16;   // score-phase A base
    const __bf16* aP = Ps + l15 * CHUNK + half * 16;       // PV-phase A base

    for (int c0 = 0; c0 < S_; c0 += CHUNK) {
        // --- scores: S = (Q K^T) * scale into LDS (one quad loop) ---------
        {
            int nloc = wave * 64;          // tiles nloc, +16, +32, +48
            long krow = (long)batch * S_ + c0 + nloc + l15;
            const __bf16* b0B = Kb + krow * D_ + half * 16;
            const __bf16* b1B = Kb + (krow + 16) * D_ + half * 16;
            const __bf16* b2B = Kb + (krow + 32) * D_ + half * 16;
            const __bf16* b3B = Kb + (krow + 48) * D_ + half * 16;
            v8f s0 = {}, s1 = {}, s2 = {}, s3 = {};
            QUAD_K_LOOP(aQ, b0B, b1B, b2B, b3B, D_, s0, s1, s2, s3);
            #pragma unroll
            for (int r = 0; r < 8; ++r) {
                float* dst = Ssc + (r + 8 * half) * CHUNK + nloc + l15;
                dst[0]  = s0[r] * SCALE;
                dst[16] = s1[r] * SCALE;
                dst[32] = s2[r] * SCALE;
                dst[48] = s3[r] * SCALE;
            }
        }
        __syncthreads();
        // --- chunk row max, online rescale factor -------------------------
        float m = -3.0e38f;
        for (int c = seg * 32; c < seg * 32 + 32; ++c)
            m = fmaxf(m, Ssc[rrow * CHUNK + c]);
        red[rrow * 16 + seg] = m;
        __syncthreads();
        if (seg == 0) {
            float cm = red[rrow * 16];
            for (int j = 1; j < 16; ++j) cm = fmaxf(cm, red[rrow * 16 + j]);
            float om = rowMax[rrow];
            float nm = fmaxf(om, cm);
            rowScale[rrow] = __expf(om - nm);
            rowMax[rrow] = nm;
            rowSum[rrow] *= rowScale[rrow];
        }
        __syncthreads();
        // --- exp into bf16 P panel + partial sums -------------------------
        float nm = rowMax[rrow], ps = 0.f;
        for (int c = seg * 32; c < seg * 32 + 32; ++c) {
            float e = __expf(Ssc[rrow * CHUNK + c] - nm);
            Ps[rrow * CHUNK + c] = (__bf16)e;
            ps += e;
        }
        red[rrow * 16 + seg] = ps;
        __syncthreads();
        if (seg == 0) {
            float s = 0.f;
            for (int j = 0; j < 16; ++j) s += red[rrow * 16 + j];
            rowSum[rrow] += s;
        }
        __syncthreads();
        // --- rescale accumulators, accumulate P.V over this chunk ---------
        float rsv[8];
        #pragma unroll
        for (int r = 0; r < 8; ++r) rsv[r] = rowScale[r + 8 * half];
        #pragma unroll
        for (int dtp = 0; dtp < 2; ++dtp) {
            int d0 = wave * 128 + dtp * 64;   // tiles d0, +16, +32, +48
            v8f a0 = accO[4 * dtp + 0], a1 = accO[4 * dtp + 1];
            v8f a2 = accO[4 * dtp + 2], a3 = accO[4 * dtp + 3];
            #pragma unroll
            for (int r = 0; r < 8; ++r) {
                a0[r] *= rsv[r]; a1[r] *= rsv[r];
                a2[r] *= rsv[r]; a3[r] *= rsv[r];
            }
            long vrow = (long)batch * D_ + d0 + l15;
            const __bf16* v0B = Vtb + vrow * S_ + c0 + half * 16;
            const __bf16* v1B = Vtb + (vrow + 16) * S_ + c0 + half * 16;
            const __bf16* v2B = Vtb + (vrow + 32) * S_ + c0 + half * 16;
            const __bf16* v3B = Vtb + (vrow + 48) * S_ + c0 + half * 16;
            QUAD_K_LOOP(aP, v0B, v1B, v2B, v3B, CHUNK, a0, a1, a2, a3);
            accO[4 * dtp + 0] = a0; accO[4 * dtp + 1] = a1;
            accO[4 * dtp + 2] = a2; accO[4 * dtp + 3] = a3;
        }
        __syncthreads();   // protect Ssc/Ps before next chunk overwrites
    }
    // --- final 1/sum scaling, store context -------------------------------
    float inv[8];
    #pragma unroll
    for (int r = 0; r < 8; ++r) inv[r] = 1.0f / rowSum[r + 8 * half];
    #pragma unroll
    for (int dt = 0; dt < 8; ++dt) {
        int d0 = wave * 128 + dt * 16;
        #pragma unroll
        for (int r = 0; r < 8; ++r)
            ctx[(q0 + r + 8 * half) * D_ + d0 + l15] = accO[dt][r] * inv[r];
    }
}

// ---------------------------------------------------------------------------
// K3: raw scaled scores for batch 1 into the weights output region.
// grid = (S/16, S/512), block = 256 (8 waves, four 16x16 tiles each).
// ---------------------------------------------------------------------------
__global__ __launch_bounds__(256) void k_scores_b1(
    const __bf16* __restrict__ Qb, const __bf16* __restrict__ Kb,
    float* __restrict__ outw) {
    const int tid = threadIdx.x, wave = tid >> 5, lane = tid & 31;
    const int half = lane >> 4, l15 = lane & 15;
    const long q0 = (long)1 * S_ + blockIdx.x * 16;     // batch 1
    const int n0 = blockIdx.y * 512 + wave * 64;        // tiles n0,+16,+32,+48
    const long krow = (long)1 * S_ + n0 + l15;
    const __bf16* aB = Qb + (q0 + l15) * D_ + half * 16;
    const __bf16* b0B = Kb + krow * D_ + half * 16;
    const __bf16* b1B = Kb + (krow + 16) * D_ + half * 16;
    const __bf16* b2B = Kb + (krow + 32) * D_ + half * 16;
    const __bf16* b3B = Kb + (krow + 48) * D_ + half * 16;
    v8f a0 = {}, a1 = {}, a2 = {}, a3 = {};
    QUAD_K_LOOP(aB, b0B, b1B, b2B, b3B, D_, a0, a1, a2, a3);
    #pragma unroll
    for (int r = 0; r < 8; ++r) {
        float* dst = outw + ((long)blockIdx.x * 16 + r + 8 * half) * S_ + n0 + l15;
        dst[0]  = a0[r] * SCALE;
        dst[16] = a1[r] * SCALE;
        dst[32] = a2[r] * SCALE;
        dst[48] = a3[r] * SCALE;
    }
}

// ---------------------------------------------------------------------------
// K4: in-place row softmax over the [S,S] weights output.
// ---------------------------------------------------------------------------
__global__ __launch_bounds__(256) void k_softmax_rows(float* __restrict__ w) {
    __shared__ float sd[256];
    float* row = w + (long)blockIdx.x * S_;
    const int tid = threadIdx.x;
    float m = -3.0e38f;
    for (int i = tid; i < S_; i += 256) m = fmaxf(m, row[i]);
    sd[tid] = m; __syncthreads();
    for (int s = 128; s > 0; s >>= 1) {
        if (tid < s) sd[tid] = fmaxf(sd[tid], sd[tid + s]);
        __syncthreads();
    }
    m = sd[0]; __syncthreads();
    float sum = 0.f;
    for (int i = tid; i < S_; i += 256) {
        float e = __expf(row[i] - m);
        row[i] = e;
        sum += e;
    }
    sd[tid] = sum; __syncthreads();
    for (int s = 128; s > 0; s >>= 1) {
        if (tid < s) sd[tid] += sd[tid + s];
        __syncthreads();
    }
    float inv = 1.0f / sd[0];
    for (int i = tid; i < S_; i += 256) row[i] *= inv;
}

// ---------------------------------------------------------------------------
extern "C" void kernel_launch(void* const* d_in, const int* in_sizes, int n_in,
                              void* d_out, int out_size, void* d_ws, size_t ws_size,
                              hipStream_t stream) {
    const float* X     = (const float*)d_in[0];
    const float* QW    = (const float*)d_in[1];
    const float* Qbias = (const float*)d_in[2];
    const float* KW    = (const float*)d_in[3];
    const float* Kbias = (const float*)d_in[4];
    const float* VW    = (const float*)d_in[5];
    const float* Vbias = (const float*)d_in[6];
    float* out_ctx = (float*)d_out;                    // [B,S,D]
    float* out_w   = out_ctx + (long)B_ * S_ * D_;     // [S,S] (batch 1 weights)

    // workspace layout (bf16): W(3*D*D) | Q | K | Vt  (~107 MB)
    __bf16* Wb  = (__bf16*)d_ws;
    __bf16* Qb  = Wb + 3L * D_ * D_;
    __bf16* Kb  = Qb + (long)MROWS * D_;
    __bf16* Vtb = Kb + (long)MROWS * D_;

    k_convert_w<<<3 * D_ * D_ / (4 * 256), 256, 0, stream>>>(QW, KW, VW, Wb);
    k_qkv<<<dim3(MROWS / 16, D_ / 512), 256, 0, stream>>>(
        X, Wb, Qbias, Kbias, Vbias, Qb, Kb, Vtb);
    k_attn<<<dim3(S_ / 16, B_), 256, 0, stream>>>(Qb, Kb, Vtb, out_ctx);
    k_scores_b1<<<dim3(S_ / 16, S_ / 512), 256, 0, stream>>>(Qb, Kb, out_w);
    k_softmax_rows<<<S_, 256, 0, stream>>>(out_w);
}
